// Generator_59596966199434
// MI455X (gfx1250) — matrix operand order repaired and checked by reading, not compiled, not run on hardware
//
#include <hip/hip_runtime.h>
#include <hip/hip_bf16.h>

#define DIMD   128
#define NPTS   512
#define NB     2
#define HPOS   64
#define HATT   512
#define KN     16

typedef __attribute__((ext_vector_type(16))) _Float16 v16h;
typedef __attribute__((ext_vector_type(8)))  float    v8f;

#define WMMA_F16(a, b, c) \
  __builtin_amdgcn_wmma_f32_16x16x32_f16(false, (a), false, (b), (short)0, (c), false, false)

// ---- WMMA fragment loaders (layouts per CDNA5 ISA 7.12.2, wave32) ----------

// A: 16x32 (MxK) f16 from LDS. lanes 0-15 -> M=0..15, K 0-7 (v0-3) / 16-23 (v4-7);
// lanes 16-31 -> same M, K 8-15 / 24-31.
__device__ __forceinline__ v16h wmma_a_lds(const _Float16* base, int ld, int lane, int kbase) {
  int m = lane & 15, half = lane >> 4;
  v16h a;
#pragma unroll
  for (int j = 0; j < 8; ++j) {
    int kp = kbase + ((j >> 2) << 4) + (half << 3) + ((j & 3) << 1);
    a[2 * j]     = base[m * ld + kp];
    a[2 * j + 1] = base[m * ld + kp + 1];
  }
  return a;
}

// B fragment from pre-packed f16 weights: per (kt,nt) tile, lane-major,
// 16 contiguous halfs per lane -> two coalesced b128 loads.
__device__ __forceinline__ v16h wmma_b_packed(const _Float16* __restrict__ P, int nn,
                                              int kt, int nt, int lane) {
  return *(const v16h*)(P + (((size_t)(kt * nn + nt) * 32 + lane) << 4));
}

// ---- Kernel 0: pack f32 weight (K x N, row-major) into f16 B-fragment order.
// Element e of lane `lane` in tile (kt,nt): k = kt*32 + (e>>3)*16 + (lane>>4)*8 + (e&7),
// col = nt*16 + (lane&15).
__global__ __launch_bounds__(256) void pack_b_kernel(const float* __restrict__ W,
                                                     _Float16* __restrict__ P,
                                                     int K, int N) {
  int t = blockIdx.x * 256 + threadIdx.x;
  if (t >= K * N) return;
  int nn = N >> 4;
  int tile = t >> 9, within = t & 511;
  int lane = within >> 4, e = within & 15;
  int kt = tile / nn, nt = tile - kt * nn;
  int k = kt * 32 + ((e >> 3) << 4) + ((lane >> 4) << 3) + (e & 7);
  int c = nt * 16 + (lane & 15);
  P[t] = (_Float16)W[(size_t)k * N + c];
}

// ---- Kernel 1: qkv = x^T @ w_qkv  (M=B*N=1024, K=128, Nc=384) --------------
// One wave per 16x16 output tile; A gathered from x (B,128,N) with on-the-fly
// f32->f16 convert (lanes 0-15 read consecutive n -> coalesced), B pre-packed.
__global__ __launch_bounds__(32) void qkv_gemm_kernel(const float* __restrict__ x,
                                                      const _Float16* __restrict__ w_qkv_p,
                                                      float* __restrict__ qkv) {
  int mt = blockIdx.x;          // 0..63
  int nt = blockIdx.y;          // 0..23
  int lane = threadIdx.x & 31;
  int m16 = lane & 15, half = lane >> 4;

  int mrow = mt * 16 + m16;     // global row = b*N + n
  int bb = mrow >> 9, n = mrow & (NPTS - 1);

  v8f acc = {};
#pragma unroll
  for (int ks = 0; ks < DIMD; ks += 32) {
    v16h a;
#pragma unroll
    for (int j = 0; j < 8; ++j) {
      int kp = ks + ((j >> 2) << 4) + (half << 3) + ((j & 3) << 1);
      a[2 * j]     = (_Float16)x[((size_t)(bb * DIMD + kp)) * NPTS + n];
      a[2 * j + 1] = (_Float16)x[((size_t)(bb * DIMD + kp + 1)) * NPTS + n];
    }
    v16h bf = wmma_b_packed(w_qkv_p, 24, ks >> 5, nt, lane);
    acc = WMMA_F16(a, bf, acc);
  }
#pragma unroll
  for (int r = 0; r < 8; ++r) {
    int m = mt * 16 + r + 8 * half;
    qkv[(size_t)m * 384 + nt * 16 + m16] = acc[r];
  }
}

// ---- Kernel 2: 16-NN indices (exact jax.lax.top_k tie-break: lower index) --
// One wave per query point; each lane owns 16 candidates, 16 rounds of
// cross-wave argmin on (dist^2, index).
__global__ __launch_bounds__(256) void topk_kernel(const float* __restrict__ pos,
                                                   int* __restrict__ idx) {
  int wave = threadIdx.x >> 5;
  int lane = threadIdx.x & 31;
  int q = blockIdx.x * 8 + wave;          // 0..B*N-1
  int b = q >> 9;

  float px = pos[(size_t)q * 3 + 0];
  float py = pos[(size_t)q * 3 + 1];
  float pz = pos[(size_t)q * 3 + 2];

  float dloc[16];
#pragma unroll
  for (int t = 0; t < 16; ++t) {
    int j = lane * 16 + t;
    float dx = px - pos[(size_t)(b * NPTS + j) * 3 + 0];
    float dy = py - pos[(size_t)(b * NPTS + j) * 3 + 1];
    float dz = pz - pos[(size_t)(b * NPTS + j) * 3 + 2];
    dloc[t] = dx * dx + dy * dy + dz * dz;   // sqrt is monotonic; ordering identical
  }

  unsigned taken = 0;
  for (int r = 0; r < KN; ++r) {
    float bd = 3.0e38f;
    int bt = -1;
#pragma unroll
    for (int t = 0; t < 16; ++t) {
      if (!((taken >> t) & 1u) && dloc[t] < bd) { bd = dloc[t]; bt = t; }  // strict < keeps lowest index
    }
    int bj = (bt >= 0) ? (lane * 16 + bt) : 0x7fffffff;
    float rd = bd;
    int rj = bj;
#pragma unroll
    for (int off = 16; off >= 1; off >>= 1) {
      float od = __shfl_xor(rd, off, 32);
      int   oj = __shfl_xor(rj, off, 32);
      if (od < rd || (od == rd && oj < rj)) { rd = od; rj = oj; }
    }
    if (bj == rj && bt >= 0) taken |= (1u << bt);   // j is globally unique -> unique winner
    if (lane == 0) idx[(size_t)q * KN + r] = rj;
  }
}

// ---- Kernel 3: fused pos-MLP + attn-MLP + softmax + aggregate --------------
// One block (8 waves) per query point. The 16 neighbors form exactly one WMMA
// M-tile; all intermediates live in LDS (~40 KB << 320 KB/WGP). Packed f16
// weights stream from L2 (total ~0.37 MB << 192 MB L2).
__global__ __launch_bounds__(256) void fused_attn_kernel(
    const float* __restrict__ pos, const float* __restrict__ qkv,
    const int* __restrict__ idx,
    const float* __restrict__ pos_w1, const float* __restrict__ pos_b1,
    const _Float16* __restrict__ pos_w2_p, const float* __restrict__ pos_b2,
    const _Float16* __restrict__ attn_w1_p, const float* __restrict__ attn_b1,
    const _Float16* __restrict__ attn_w2_p, const float* __restrict__ attn_b2,
    float* __restrict__ out) {
  __shared__ int      s_idx[KN];
  __shared__ float    s_rel[KN][3];
  __shared__ _Float16 s_ph[KN][HPOS + 8];     // relu(rel_pos @ pos_w1 + b1), f16
  __shared__ float    s_pe[KN][DIMD];         // pos embedding (f32, reused twice)
  __shared__ _Float16 s_x[KN][DIMD + 8];      // q - k_gathered + pe, f16
  __shared__ _Float16 s_h[KN][HATT + 8];      // relu(layer1), f16
  __shared__ float    s_s[KN][DIMD];          // sim (pre-softmax)

  int i = blockIdx.x, b = blockIdx.y;
  int tid = threadIdx.x;
  int wave = tid >> 5, lane = tid & 31;
  int q = b * NPTS + i;

  // Phase 0: gather indices, relative positions
  if (tid < KN) {
    int j = idx[(size_t)q * KN + tid];
    s_idx[tid] = j;
#pragma unroll
    for (int c = 0; c < 3; ++c)
      s_rel[tid][c] = pos[(size_t)q * 3 + c] - pos[(size_t)(b * NPTS + j) * 3 + c];
  }
  __syncthreads();

  // Phase 1: pos hidden = relu(rel @ pos_w1 + b1)  (K=3 -> plain VALU)
  for (int e = tid; e < KN * HPOS; e += 256) {
    int m = e >> 6, c = e & (HPOS - 1);
    float acc = pos_b1[c];
    acc += s_rel[m][0] * pos_w1[0 * HPOS + c];
    acc += s_rel[m][1] * pos_w1[1 * HPOS + c];
    acc += s_rel[m][2] * pos_w1[2 * HPOS + c];
    s_ph[m][c] = (_Float16)fmaxf(acc, 0.0f);
  }
  __syncthreads();

  // Phase 2: pe = hidden(16x64) @ pos_w2(64x128) + b2 ; one 16x16 N-tile/wave
  {
    v8f acc = {};
#pragma unroll
    for (int ks = 0; ks < HPOS; ks += 32) {
      v16h a = wmma_a_lds(&s_ph[0][0], HPOS + 8, lane, ks);
      v16h bf = wmma_b_packed(pos_w2_p, 8, ks >> 5, wave, lane);
      acc = WMMA_F16(a, bf, acc);
    }
    int n = wave * 16 + (lane & 15), half = lane >> 4;
#pragma unroll
    for (int r = 0; r < 8; ++r) s_pe[r + 8 * half][n] = acc[r] + pos_b2[n];
  }
  __syncthreads();

  // Phase 3: X = q_i - k_gathered + pe  (f16)
  {
    const float* qrow = qkv + (size_t)q * 384;
    for (int e = tid; e < KN * DIMD; e += 256) {
      int m = e >> 7, d = e & (DIMD - 1);
      const float* krow = qkv + (size_t)(b * NPTS + s_idx[m]) * 384 + DIMD;
      s_x[m][d] = (_Float16)(qrow[d] - krow[d] + s_pe[m][d]);
    }
  }
  __syncthreads();

  // Phase 4: H = relu(X(16x128) @ attn_w1(128x512) + b1) ; 4 N-tiles/wave.
  // A fragments hoisted: reused across all 4 N-tiles from VGPRs.
  {
    v16h a0 = wmma_a_lds(&s_x[0][0], DIMD + 8, lane, 0);
    v16h a1 = wmma_a_lds(&s_x[0][0], DIMD + 8, lane, 32);
    v16h a2 = wmma_a_lds(&s_x[0][0], DIMD + 8, lane, 64);
    v16h a3 = wmma_a_lds(&s_x[0][0], DIMD + 8, lane, 96);
#pragma unroll
    for (int t = 0; t < 4; ++t) {
      int nt = wave * 4 + t;
      __builtin_prefetch(attn_w1_p + (((size_t)(0 * 32 + nt + 1) * 32 + lane) << 4), 0, 3);
      v8f acc = {};
      acc = WMMA_F16(a0, wmma_b_packed(attn_w1_p, 32, 0, nt, lane), acc);
      acc = WMMA_F16(a1, wmma_b_packed(attn_w1_p, 32, 1, nt, lane), acc);
      acc = WMMA_F16(a2, wmma_b_packed(attn_w1_p, 32, 2, nt, lane), acc);
      acc = WMMA_F16(a3, wmma_b_packed(attn_w1_p, 32, 3, nt, lane), acc);
      int n = nt * 16 + (lane & 15), half = lane >> 4;
#pragma unroll
      for (int r = 0; r < 8; ++r)
        s_h[r + 8 * half][n] = (_Float16)fmaxf(acc[r] + attn_b1[n], 0.0f);
    }
  }
  __syncthreads();

  // Phase 5: S = H(16x512) @ attn_w2(512x128) + b2 ; 1 N-tile/wave, 16 K-steps
  {
    v8f acc = {};
    for (int ks = 0; ks < HATT; ks += 32) {
      if (ks + 32 < HATT)
        __builtin_prefetch(attn_w2_p + (((size_t)(((ks >> 5) + 1) * 8 + wave) * 32 + lane) << 4), 0, 3);
      v16h a = wmma_a_lds(&s_h[0][0], HATT + 8, lane, ks);
      v16h bf = wmma_b_packed(attn_w2_p, 8, ks >> 5, wave, lane);
      acc = WMMA_F16(a, bf, acc);
    }
    int n = wave * 16 + (lane & 15), half = lane >> 4;
#pragma unroll
    for (int r = 0; r < 8; ++r) s_s[r + 8 * half][n] = acc[r] + attn_b2[n];
  }
  __syncthreads();

  // Phase 6: softmax over the 16 neighbors (per channel) + weighted aggregate
  if (tid < DIMD) {
    int d = tid;
    float mx = -3.0e38f;
#pragma unroll
    for (int m = 0; m < KN; ++m) mx = fmaxf(mx, s_s[m][d]);
    float ev[KN];
    float sum = 0.0f;
#pragma unroll
    for (int m = 0; m < KN; ++m) { ev[m] = __expf(s_s[m][d] - mx); sum += ev[m]; }
    float inv = 1.0f / sum;
    float agg = 0.0f;
#pragma unroll
    for (int m = 0; m < KN; ++m) {
      float vg = qkv[(size_t)(b * NPTS + s_idx[m]) * 384 + 2 * DIMD + d] + s_pe[m][d];
      agg += ev[m] * inv * vg;
    }
    out[(size_t)(b * DIMD + d) * NPTS + i] = agg;   // (B, DIM, N)
  }
}

// ---- Launcher ---------------------------------------------------------------
extern "C" void kernel_launch(void* const* d_in, const int* in_sizes, int n_in,
                              void* d_out, int out_size, void* d_ws, size_t ws_size,
                              hipStream_t stream) {
  const float* x       = (const float*)d_in[0];   // (B,128,N)
  const float* pos     = (const float*)d_in[1];   // (B,N,3)
  const float* w_qkv   = (const float*)d_in[2];   // (128,384)
  const float* pos_w1  = (const float*)d_in[3];   // (3,64)
  const float* pos_b1  = (const float*)d_in[4];   // (64,)
  const float* pos_w2  = (const float*)d_in[5];   // (64,128)
  const float* pos_b2  = (const float*)d_in[6];   // (128,)
  const float* attn_w1 = (const float*)d_in[7];   // (128,512)
  const float* attn_b1 = (const float*)d_in[8];   // (512,)
  const float* attn_w2 = (const float*)d_in[9];   // (512,128)
  const float* attn_b2 = (const float*)d_in[10];  // (128,)
  float* out = (float*)d_out;

  // Workspace layout (all 32B-aligned):
  //   qkv  (B*N,384) f32           @ 0          (1,572,864 B)
  //   idx  (B*N,16)  i32           @ 1,572,864  (   65,536 B)
  //   packed f16 weights           @ 1,638,400  (  376,832 B)
  char* ws = (char*)d_ws;
  float*     qkv       = (float*)ws;
  int*       idx       = (int*)(ws + 1572864);
  _Float16*  w_qkv_p   = (_Float16*)(ws + 1638400);              // 128*384 = 49152 h
  _Float16*  pos_w2_p  = w_qkv_p + 128 * 384;                    // 64*128  = 8192 h
  _Float16*  attn_w1_p = pos_w2_p + 64 * 128;                    // 128*512 = 65536 h
  _Float16*  attn_w2_p = attn_w1_p + 128 * 512;                  // 512*128 = 65536 h

  // 0) Pack weights into WMMA B-fragment order (f16)
  pack_b_kernel<<<(128 * 384 + 255) / 256, 256, 0, stream>>>(w_qkv,   w_qkv_p,   128, 384);
  pack_b_kernel<<<(64 * 128 + 255) / 256, 256, 0, stream>>>(pos_w2,  pos_w2_p,  64, 128);
  pack_b_kernel<<<(128 * 512 + 255) / 256, 256, 0, stream>>>(attn_w1, attn_w1_p, 128, 512);
  pack_b_kernel<<<(512 * 128 + 255) / 256, 256, 0, stream>>>(attn_w2, attn_w2_p, 512, 128);

  // 1) QKV projection: 64 M-tiles x 24 N-tiles, one wave each
  qkv_gemm_kernel<<<dim3(64, 24), 32, 0, stream>>>(x, w_qkv_p, qkv);

  // 2) 16-NN selection: 1024 queries, 8 waves/block
  topk_kernel<<<dim3((NB * NPTS) / 8), 256, 0, stream>>>(pos, idx);

  // 3) Fused pos-MLP + attention MLP + softmax + aggregation: one block/query
  fused_attn_kernel<<<dim3(NPTS, NB), 256, 0, stream>>>(
      pos, qkv, idx, pos_w1, pos_b1, pos_w2_p, pos_b2,
      attn_w1_p, attn_b1, attn_w2_p, attn_b2, out);

  (void)in_sizes; (void)n_in; (void)out_size; (void)ws_size;
}
